// GNN_28063316312188
// MI455X (gfx1250) — matrix operand order repaired and checked by reading, not compiled, not run on hardware
//
#include <hip/hip_runtime.h>

// ---------------------------------------------------------------------------
// CDNA5 (gfx1250) fused GNN pipeline, wave32 + v_wmma_f32_16x16x32_bf16.
// D = H + 6 = 256 exactly; gates padded 750 -> 3*256.
// ---------------------------------------------------------------------------

typedef __bf16 bf16;
typedef __attribute__((ext_vector_type(16))) __bf16 v16bf;
typedef __attribute__((ext_vector_type(8)))  float  v8f;

// global (address_space 1) pointer to a B fragment chunk: forces
// global_load_b128 (LOADcnt only) instead of flat_load_b128 (LOAD+DScnt)
// even when the base pointer has been made opaque by an asm barrier.
typedef const __attribute__((address_space(1))) v16bf* gv16bf_ptr;

__device__ inline v8f vzero8() {
    v8f v;
    #pragma unroll
    for (int i = 0; i < 8; ++i) v[i] = 0.f;
    return v;
}

__device__ inline v8f wmma_bf16(v16bf a, v16bf b, v8f c) {
    // D = A(16x32 bf16) * B(32x16 bf16) + C(16x16 f32)
    return __builtin_amdgcn_wmma_f32_16x16x32_bf16(false, a, false, b,
                                                   (short)0, c, false, false);
}

// A fragment (16x32, bf16) from LDS.  Per ISA: lanes 0-15 hold rows M=0..15
// with K halves {0..7,16..23}; lanes 16-31 same rows, K halves {8..15,24..31}.
// Element e carries K = (e&7) + (e>=8 ? 16 : 0) + (lane<16 ? 0 : 8); K-pairs
// are contiguous -> eight 32-bit DS loads per fragment (merged to 2addr_b64).
union AFrag { v16bf v; unsigned u[8]; };

__device__ inline v16bf loadA(const bf16* lds, int stride, int mbase, int kbase, int lane) {
    AFrag f;
    const int row = mbase + (lane & 15);
    const unsigned* rp = (const unsigned*)(lds + (size_t)row * stride);
    const int k8 = (lane < 16) ? 0 : 8;
    #pragma unroll
    for (int v = 0; v < 8; ++v) {
        int k = ((v & 4) ? 16 : 0) + k8 + 2 * (v & 3);
        f.u[v] = rp[(kbase + k) >> 1];
    }
    return f.v;
}

// B fragments pre-swizzled in global memory: fragment f = kstep*NT + ntile,
// each lane owns 16 consecutive bf16 (32B) -> two global_load_b128 per frag.
__device__ inline v16bf loadB(const bf16* pk, int NT, int kstep, int ntile, int lane) {
    const v16bf* p = (const v16bf*)pk + ((size_t)(kstep * NT + ntile) * 32 + lane);
    return *(gv16bf_ptr)p;
}

__device__ inline float sigmoidf_(float x) { return 1.f / (1.f + __expf(-x)); }

// ---------------------------------------------------------------------------
// Weight packing: rearrange an f32 weight into bf16 WMMA-B fragment order.
// ---------------------------------------------------------------------------
__global__ void k_pack(const float* __restrict__ W, bf16* __restrict__ out,
                       int Kpad, int Npad, int kReal, int nRealPerBlock,
                       int nBlockPad, int ldW, int transpose) {
    int idx = blockIdx.x * 256 + threadIdx.x;
    int total = Kpad * Npad;
    if (idx >= total) return;
    int k = idx / Npad, n = idx - k * Npad;
    int blk = n / nBlockPad, j = n - blk * nBlockPad;
    float val = 0.f;
    if (j < nRealPerBlock && k < kReal) {
        int nr = blk * nRealPerBlock + j;
        val = transpose ? W[(size_t)nr * ldW + k] : W[(size_t)k * ldW + nr];
    }
    int f    = (k >> 5) * (Npad >> 4) + (n >> 4);
    int lane = (n & 15) + (((k >> 3) & 1) << 4);
    int e    = (k & 7) + (((k >> 4) & 1) << 3);
    out[((size_t)(f * 32 + lane)) * 16 + e] = (bf16)val;
}

// Gate biases, padded to 4 x 256: [r_sum, z_sum, in(bih_n), hn(bhh_n)]
__global__ void k_pack_gbias(const float* __restrict__ bih,
                             const float* __restrict__ bhh,
                             float* __restrict__ gb) {
    int c = threadIdx.x;
    float rs = 0.f, zs = 0.f, in = 0.f, hn = 0.f;
    if (c < 250) {
        rs = bih[c] + bhh[c];
        zs = bih[250 + c] + bhh[250 + c];
        in = bih[500 + c];
        hn = bhh[500 + c];
    }
    gb[0 * 256 + c] = rs; gb[1 * 256 + c] = zs;
    gb[2 * 256 + c] = in; gb[3 * 256 + c] = hn;
}

__global__ void k_zero_counters(int* cnts) {
    if (threadIdx.x < 2) cnts[threadIdx.x] = 0;
}

// feats = [one-hot(6) | hid(250)] bf16; build compacted map/mem node lists.
__global__ void k_init_nodes(const int* __restrict__ type_idx,
                             bf16* __restrict__ feats,
                             int* __restrict__ idx_map, int* __restrict__ idx_mem,
                             int* __restrict__ cnts, int N) {
    int i = blockIdx.x * 256 + threadIdx.x;
    if (i >= N) return;
    int t = type_idx[i];
    bf16* row = feats + (size_t)i * 256;
    #pragma unroll
    for (int c = 0; c < 6; ++c) row[c] = (bf16)(c == t ? 1.f : 0.f);
    for (int c = 6; c < 256; ++c) row[c] = (bf16)0.f;
    if (t == 0)      idx_map[atomicAdd(&cnts[0], 1)] = i;
    else if (t == 5) idx_mem[atomicAdd(&cnts[1], 1)] = i;
}

// agg init: monotonic encoding of -inf  (~0xFF800000 = 0x007FFFFF)
__global__ void k_fill_agg(unsigned* __restrict__ agg, unsigned long long total) {
    unsigned long long i = (unsigned long long)blockIdx.x * 256 + threadIdx.x;
    if (i < total) agg[i] = 0x007FFFFFu;
}

// ---------------------------------------------------------------------------
// Fused GRU over a compacted node list.  One workgroup = 32 nodes, 16 steps,
// hidden state in LDS (bf16, double-buffered).  The two hidden column-tiles
// per wave are processed sequentially (8 accumulator tiles live).  Weight
// base pointers are made opaque per step with an empty asm barrier so the
// compiler re-fetches B fragments from L2 each step instead of hoisting all
// of them out of the recurrence and spilling to scratch.
// Gate columns padded: r -> [0,256), z -> [256,512), n -> [512,768).
// ---------------------------------------------------------------------------
__global__ void __launch_bounds__(256, 1)
k_gru(const int* __restrict__ list, const int* __restrict__ cnts, int cntSel,
      const int* __restrict__ attr, const float* __restrict__ emb,
      const bf16* __restrict__ WihPk, const bf16* __restrict__ WhhPk,
      const float* __restrict__ gbias, bf16* __restrict__ feats) {
    __shared__ bf16 sH[2][32][268];   // stride 268: bank-conflict-free A reads
    __shared__ bf16 sX[32][76];
    __shared__ int  sNode[32];
    __shared__ int  sTok[32 * 16];

    const int cnt = cnts[cntSel];
    const int tile = blockIdx.x * 32;
    if (tile >= cnt) return;
    const int tid = threadIdx.x;
    const int wave = tid >> 5, lane = tid & 31;

    if (tid < 32) {
        int g = tile + tid;
        sNode[tid] = list[g < cnt ? g : cnt - 1];
    }
    __syncthreads();
    for (int i = tid; i < 32 * 16; i += 256)
        sTok[i] = attr[(size_t)sNode[i >> 4] * 16 + (i & 15)];
    for (int i = tid; i < 32 * 268; i += 256)
        ((bf16*)sH[0])[i] = (bf16)0.f;
    __syncthreads();

    // hoist gate biases (wave/lane-invariant across all 16 steps)
    float bias_r[2], bias_z[2], bias_i[2], bias_h[2];
    #pragma unroll
    for (int c = 0; c < 2; ++c) {
        int ncol = (wave * 2 + c) * 16 + (lane & 15);
        bias_r[c] = gbias[0 * 256 + ncol];
        bias_z[c] = gbias[1 * 256 + ncol];
        bias_i[c] = gbias[2 * 256 + ncol];
        bias_h[c] = gbias[3 * 256 + ncol];
    }

    const bf16* wih = WihPk;
    const bf16* whh = WhhPk;

    int cur = 0;
    for (int t = 0; t < 16; ++t) {
        // Opaque per-iteration redefinition of the weight pointers: prevents
        // LICM from hoisting ~60 B-fragment loads out of the 16-step loop
        // (which previously spilled them to scratch).  L2 serves them instead.
        asm volatile("" : "+s"(wih), "+s"(whh));

        {   // gather token embeddings for this step -> sX (bf16)
            int row = tid >> 3, part = (tid & 7) * 8;
            const float* er = emb + (size_t)sTok[row * 16 + t] * 64 + part;
            bf16* xr = &sX[row][part];
            #pragma unroll
            for (int q = 0; q < 8; ++q) xr[q] = (bf16)er[q];
        }
        __syncthreads();

        #pragma unroll
        for (int c = 0; c < 2; ++c) {          // sequential column-tiles
            const int ct = wave * 2 + c;
            v8f acc[2][4];                      // [Mtile][kind: r,z,in,hn]
            #pragma unroll
            for (int a = 0; a < 2; ++a)
                #pragma unroll
                for (int k = 0; k < 4; ++k) acc[a][k] = vzero8();

            // input contribution: K = 64 (2 k-steps)
            #pragma unroll
            for (int ks = 0; ks < 2; ++ks) {
                v16bf a0 = loadA(&sX[0][0], 76, 0, ks * 32, lane);
                v16bf a1 = loadA(&sX[0][0], 76, 16, ks * 32, lane);
                #pragma unroll
                for (int g = 0; g < 3; ++g) {
                    v16bf b = loadB(wih, 48, ks, ct + g * 16, lane);
                    acc[0][g] = wmma_bf16(a0, b, acc[0][g]);
                    acc[1][g] = wmma_bf16(a1, b, acc[1][g]);
                }
            }
            // hidden contribution: K = 256 (8 k-steps); n-gate kept separate
            #pragma unroll
            for (int ks = 0; ks < 8; ++ks) {
                v16bf a0 = loadA(&sH[cur][0][0], 268, 0, ks * 32, lane);
                v16bf a1 = loadA(&sH[cur][0][0], 268, 16, ks * 32, lane);
                #pragma unroll
                for (int g = 0; g < 3; ++g) {
                    v16bf b = loadB(whh, 48, ks, ct + g * 16, lane);
                    int kind = (g == 2) ? 3 : g;
                    acc[0][kind] = wmma_bf16(a0, b, acc[0][kind]);
                    acc[1][kind] = wmma_bf16(a1, b, acc[1][kind]);
                }
            }
            // gates + state update -> other LDS buffer
            const int ncol = ct * 16 + (lane & 15);
            #pragma unroll
            for (int mt = 0; mt < 2; ++mt) {
                #pragma unroll
                for (int rr = 0; rr < 8; ++rr) {
                    int row = mt * 16 + rr + ((lane < 16) ? 0 : 8);
                    float r = sigmoidf_(acc[mt][0][rr] + bias_r[c]);
                    float z = sigmoidf_(acc[mt][1][rr] + bias_z[c]);
                    float n = tanhf(acc[mt][2][rr] + bias_i[c] +
                                    r * (acc[mt][3][rr] + bias_h[c]));
                    float hold = (float)sH[cur][row][ncol];
                    sH[cur ^ 1][row][ncol] = (bf16)((1.f - z) * n + z * hold);
                }
            }
        }
        __syncthreads();
        cur ^= 1;
    }
    // scatter hidden state into feats[:, 6:256]
    for (int i = tid; i < 32 * 250; i += 256) {
        int row = i / 250, c = i - row * 250;
        if (tile + row < cnt)
            feats[(size_t)sNode[row] * 256 + 6 + c] = sH[cur][row][c];
    }
}

// ---------------------------------------------------------------------------
// Edge message GEMM + scatter-max.  64 gathered rows per WG, 256x256 weight,
// monotonic-uint atomicMax aggregation (exact f32 max, order-independent).
// ---------------------------------------------------------------------------
__global__ void __launch_bounds__(256, 1)
k_msg(const bf16* __restrict__ feats, const bf16* __restrict__ Wpk,
      const float* __restrict__ bias, const int* __restrict__ gidx,
      const int* __restrict__ sidx, unsigned* __restrict__ agg, int rows) {
    __shared__ bf16 sA[64][268];
    __shared__ int  sScat[64];
    const int tile = blockIdx.x * 64;
    const int tid = threadIdx.x, wave = tid >> 5, lane = tid & 31;
    {
        int row = tid >> 2, part = (tid & 3) * 64;
        int g = tile + row;
        int gc = g < rows ? g : rows - 1;
        int sr = gidx ? gidx[gc] : gc;
        const unsigned* sp = (const unsigned*)(feats + (size_t)sr * 256 + part);
        unsigned* dp = (unsigned*)&sA[row][part];
        #pragma unroll
        for (int q = 0; q < 32; ++q) dp[q] = sp[q];
    }
    if (tid < 64) {
        int g = tile + tid;
        int gc = g < rows ? g : rows - 1;
        sScat[tid] = sidx ? sidx[gc] : gc;
    }
    __syncthreads();

    v8f acc[4][2];
    #pragma unroll
    for (int mt = 0; mt < 4; ++mt) { acc[mt][0] = vzero8(); acc[mt][1] = vzero8(); }
    #pragma unroll
    for (int ks = 0; ks < 8; ++ks) {
        v16bf b0 = loadB(Wpk, 16, ks, wave * 2 + 0, lane);
        v16bf b1 = loadB(Wpk, 16, ks, wave * 2 + 1, lane);
        #pragma unroll
        for (int mt = 0; mt < 4; ++mt) {
            v16bf a = loadA(&sA[0][0], 268, mt * 16, ks * 32, lane);
            acc[mt][0] = wmma_bf16(a, b0, acc[mt][0]);
            acc[mt][1] = wmma_bf16(a, b1, acc[mt][1]);
        }
    }
    #pragma unroll
    for (int c = 0; c < 2; ++c) {
        int ncol = (wave * 2 + c) * 16 + (lane & 15);
        float bv = bias[ncol];
        #pragma unroll
        for (int mt = 0; mt < 4; ++mt) {
            #pragma unroll
            for (int rr = 0; rr < 8; ++rr) {
                int row = mt * 16 + rr + ((lane < 16) ? 0 : 8);
                if (tile + row < rows) {
                    float v = acc[mt][c][rr] + bv;
                    int iv = __float_as_int(v);
                    unsigned mono = (unsigned)iv ^ (unsigned)((iv >> 31) | (int)0x80000000);
                    atomicMax(&agg[(size_t)sScat[row] * 256 + ncol], mono);
                }
            }
        }
    }
}

// decode agg (mono -> f32, non-finite -> 0), result = agg @ Wout + bout  (bf16)
__global__ void __launch_bounds__(256, 1)
k_result(const unsigned* __restrict__ agg, const bf16* __restrict__ Wpk,
         const float* __restrict__ bias, bf16* __restrict__ resultbf, int N) {
    __shared__ bf16 sA[64][268];
    const int tile = blockIdx.x * 64;
    const int tid = threadIdx.x, wave = tid >> 5, lane = tid & 31;
    {
        int row = tid >> 2, part = (tid & 3) * 64;
        int g = tile + row;
        int gc = g < N ? g : N - 1;
        const unsigned* ap = agg + (size_t)gc * 256 + part;
        bf16* dp = &sA[row][part];
        #pragma unroll
        for (int q = 0; q < 64; ++q) {
            unsigned m = ap[q];
            int iv = (m & 0x80000000u) ? (int)(m ^ 0x80000000u) : (int)~m;
            float f = __int_as_float(iv);
            if (!(f == f) || fabsf(f) > 3.0e38f) f = 0.f;
            dp[q] = (bf16)f;
        }
    }
    __syncthreads();

    v8f acc[4][2];
    #pragma unroll
    for (int mt = 0; mt < 4; ++mt) { acc[mt][0] = vzero8(); acc[mt][1] = vzero8(); }
    #pragma unroll
    for (int ks = 0; ks < 8; ++ks) {
        v16bf b0 = loadB(Wpk, 16, ks, wave * 2 + 0, lane);
        v16bf b1 = loadB(Wpk, 16, ks, wave * 2 + 1, lane);
        #pragma unroll
        for (int mt = 0; mt < 4; ++mt) {
            v16bf a = loadA(&sA[0][0], 268, mt * 16, ks * 32, lane);
            acc[mt][0] = wmma_bf16(a, b0, acc[mt][0]);
            acc[mt][1] = wmma_bf16(a, b1, acc[mt][1]);
        }
    }
    #pragma unroll
    for (int c = 0; c < 2; ++c) {
        int ncol = (wave * 2 + c) * 16 + (lane & 15);
        float bv = bias[ncol];
        #pragma unroll
        for (int mt = 0; mt < 4; ++mt) {
            #pragma unroll
            for (int rr = 0; rr < 8; ++rr) {
                int row = mt * 16 + rr + ((lane < 16) ? 0 : 8);
                int g = tile + row;
                if (g < N)
                    resultbf[(size_t)g * 256 + ncol] = (bf16)(acc[mt][c][rr] + bv);
            }
        }
    }
}

// heads: h1 = relu(result[gather] @ W1 + b1); out = (h1 @ W2 + b2)[:, :2]
__global__ void __launch_bounds__(256, 1)
k_head(const bf16* __restrict__ resultbf, const int* __restrict__ gidx,
       const bf16* __restrict__ W1pk, const float* __restrict__ b1,
       const bf16* __restrict__ W2pk, const float* __restrict__ b2,
       float* __restrict__ out, int K) {
    __shared__ bf16 sA[64][268];
    const int tile = blockIdx.x * 64;
    const int tid = threadIdx.x, wave = tid >> 5, lane = tid & 31;
    {
        int row = tid >> 2, part = (tid & 3) * 64;
        int g = tile + row;
        int gc = g < K ? g : K - 1;
        int sr = gidx[gc];
        const unsigned* sp = (const unsigned*)(resultbf + (size_t)sr * 256 + part);
        unsigned* dp = (unsigned*)&sA[row][part];
        #pragma unroll
        for (int q = 0; q < 32; ++q) dp[q] = sp[q];
    }
    __syncthreads();

    v8f acc[4][2];
    #pragma unroll
    for (int mt = 0; mt < 4; ++mt) { acc[mt][0] = vzero8(); acc[mt][1] = vzero8(); }
    #pragma unroll
    for (int ks = 0; ks < 8; ++ks) {
        v16bf b0 = loadB(W1pk, 16, ks, wave * 2 + 0, lane);
        v16bf b1f = loadB(W1pk, 16, ks, wave * 2 + 1, lane);
        #pragma unroll
        for (int mt = 0; mt < 4; ++mt) {
            v16bf a = loadA(&sA[0][0], 268, mt * 16, ks * 32, lane);
            acc[mt][0] = wmma_bf16(a, b0, acc[mt][0]);
            acc[mt][1] = wmma_bf16(a, b1f, acc[mt][1]);
        }
    }
    __syncthreads();   // all reads of sA done; reuse it for h1
    #pragma unroll
    for (int c = 0; c < 2; ++c) {
        int ncol = (wave * 2 + c) * 16 + (lane & 15);
        float bv = b1[ncol];
        #pragma unroll
        for (int mt = 0; mt < 4; ++mt) {
            #pragma unroll
            for (int rr = 0; rr < 8; ++rr) {
                int row = mt * 16 + rr + ((lane < 16) ? 0 : 8);
                float v = acc[mt][c][rr] + bv;
                sA[row][ncol] = (bf16)(v > 0.f ? v : 0.f);
            }
        }
    }
    __syncthreads();
    if (wave < 4) {   // GEMM2: [64,256] x [256,16-pad], only cols 0..1 real
        v8f a2 = vzero8();
        #pragma unroll
        for (int ks = 0; ks < 8; ++ks) {
            v16bf b = loadB(W2pk, 1, ks, 0, lane);
            v16bf a = loadA(&sA[0][0], 268, wave * 16, ks * 32, lane);
            a2 = wmma_bf16(a, b, a2);
        }
        int ncol = lane & 15;
        if (ncol < 2) {
            #pragma unroll
            for (int rr = 0; rr < 8; ++rr) {
                int row = wave * 16 + rr + ((lane < 16) ? 0 : 8);
                int g = tile + row;
                if (g < K) out[(size_t)g * 2 + ncol] = a2[rr] + b2[ncol];
            }
        }
    }
}

// ---------------------------------------------------------------------------
extern "C" void kernel_launch(void* const* d_in, const int* in_sizes, int n_in,
                              void* d_out, int out_size, void* d_ws, size_t ws_size,
                              hipStream_t stream) {
    const int*   attr     = (const int*)  d_in[0];
    const int*   type_idx = (const int*)  d_in[1];
    const int*   src      = (const int*)  d_in[2];
    const int*   dst      = (const int*)  d_in[3];
    const int*   tgt_task = (const int*)  d_in[4];
    const int*   tgt_map  = (const int*)  d_in[5];
    const float* emb      = (const float*)d_in[6];
    const float* Wih_map  = (const float*)d_in[7];
    const float* Whh_map  = (const float*)d_in[8];
    const float* bih_map  = (const float*)d_in[9];
    const float* bhh_map  = (const float*)d_in[10];
    const float* Wih_mem  = (const float*)d_in[11];
    const float* Whh_mem  = (const float*)d_in[12];
    const float* bih_mem  = (const float*)d_in[13];
    const float* bhh_mem  = (const float*)d_in[14];
    const float* Wf   = (const float*)d_in[15]; const float* bfb  = (const float*)d_in[16];
    const float* Wb   = (const float*)d_in[17]; const float* bbb  = (const float*)d_in[18];
    const float* Wsw  = (const float*)d_in[19]; const float* bsb  = (const float*)d_in[20];
    const float* Wout = (const float*)d_in[21]; const float* bout = (const float*)d_in[22];
    const float* Wt1  = (const float*)d_in[23]; const float* bt1  = (const float*)d_in[24];
    const float* Wt2  = (const float*)d_in[25]; const float* bt2  = (const float*)d_in[26];
    const float* Wm1  = (const float*)d_in[27]; const float* bm1  = (const float*)d_in[28];
    const float* Wm2  = (const float*)d_in[29]; const float* bm2  = (const float*)d_in[30];

    const int N  = in_sizes[1];
    const int E  = in_sizes[2];
    const int KT = in_sizes[4];
    const int KM = in_sizes[5];

    char* ws = (char*)d_ws;
    size_t off = 0;
    auto take = [&](size_t bytes) -> void* {
        off = (off + 255) & ~(size_t)255;
        void* p = ws + off;
        off += bytes;
        return p;
    };

    bf16*     feats    = (bf16*)    take((size_t)N * 256 * 2);
    unsigned* agg      = (unsigned*)take((size_t)N * 256 * 4);
    bf16*     resultbf = (bf16*)    take((size_t)N * 256 * 2);
    bf16* WihPk0 = (bf16*)take((size_t)64  * 768 * 2);
    bf16* WihPk1 = (bf16*)take((size_t)64  * 768 * 2);
    bf16* WhhPk0 = (bf16*)take((size_t)256 * 768 * 2);
    bf16* WhhPk1 = (bf16*)take((size_t)256 * 768 * 2);
    bf16* WfPk   = (bf16*)take((size_t)256 * 256 * 2);
    bf16* WbPk   = (bf16*)take((size_t)256 * 256 * 2);
    bf16* WsPk   = (bf16*)take((size_t)256 * 256 * 2);
    bf16* WoutPk = (bf16*)take((size_t)256 * 256 * 2);
    bf16* Wt1Pk[2], *Wm1Pk[2], *Wt2Pk[2], *Wm2Pk[2];
    for (int t = 0; t < 2; ++t) Wt1Pk[t] = (bf16*)take((size_t)256 * 256 * 2);
    for (int t = 0; t < 2; ++t) Wm1Pk[t] = (bf16*)take((size_t)256 * 256 * 2);
    for (int t = 0; t < 2; ++t) Wt2Pk[t] = (bf16*)take((size_t)256 * 16 * 2);
    for (int t = 0; t < 2; ++t) Wm2Pk[t] = (bf16*)take((size_t)256 * 16 * 2);
    float* gb0 = (float*)take(4 * 256 * 4);
    float* gb1 = (float*)take(4 * 256 * 4);
    int* idx_map = (int*)take((size_t)N * 4);
    int* idx_mem = (int*)take((size_t)N * 4);
    int* cnts    = (int*)take(256);

    auto cdiv = [](long long a, long long b) { return (int)((a + b - 1) / b); };

    // --- init + lists + agg fill -------------------------------------------
    k_zero_counters<<<1, 64, 0, stream>>>(cnts);
    k_init_nodes<<<cdiv(N, 256), 256, 0, stream>>>(type_idx, feats, idx_map, idx_mem, cnts, N);
    k_fill_agg<<<cdiv((long long)N * 256, 256), 256, 0, stream>>>(agg, (unsigned long long)N * 256);

    // --- pack weights to WMMA fragment order (bf16) ------------------------
    #define PACK(Wsrc, dstp, Kp, Np, kR, nRB, nBP, ld, tr) \
        k_pack<<<cdiv((long long)(Kp) * (Np), 256), 256, 0, stream>>>(Wsrc, dstp, Kp, Np, kR, nRB, nBP, ld, tr)
    PACK(Wih_map, WihPk0, 64, 768, 64, 250, 256, 64, 1);
    PACK(Wih_mem, WihPk1, 64, 768, 64, 250, 256, 64, 1);
    PACK(Whh_map, WhhPk0, 256, 768, 250, 250, 256, 250, 1);
    PACK(Whh_mem, WhhPk1, 256, 768, 250, 250, 256, 250, 1);
    PACK(Wf,   WfPk,   256, 256, 256, 256, 256, 256, 0);
    PACK(Wb,   WbPk,   256, 256, 256, 256, 256, 256, 0);
    PACK(Wsw,  WsPk,   256, 256, 256, 256, 256, 256, 0);
    PACK(Wout, WoutPk, 256, 256, 256, 256, 256, 256, 0);
    for (int t = 0; t < 2; ++t) {
        PACK(Wt1 + (size_t)t * 256 * 256, Wt1Pk[t], 256, 256, 256, 256, 256, 256, 0);
        PACK(Wm1 + (size_t)t * 256 * 256, Wm1Pk[t], 256, 256, 256, 256, 256, 256, 0);
        PACK(Wt2 + (size_t)t * 256 * 2,   Wt2Pk[t], 256, 16, 256, 2, 16, 2, 0);
        PACK(Wm2 + (size_t)t * 256 * 2,   Wm2Pk[t], 256, 16, 256, 2, 16, 2, 0);
    }
    #undef PACK
    k_pack_gbias<<<1, 256, 0, stream>>>(bih_map, bhh_map, gb0);
    k_pack_gbias<<<1, 256, 0, stream>>>(bih_mem, bhh_mem, gb1);

    // --- fused per-type GRUs (only map/mem nodes) --------------------------
    k_gru<<<cdiv(N, 32), 256, 0, stream>>>(idx_map, cnts, 0, attr, emb, WihPk0, WhhPk0, gb0, feats);
    k_gru<<<cdiv(N, 32), 256, 0, stream>>>(idx_mem, cnts, 1, attr, emb, WihPk1, WhhPk1, gb1, feats);

    // --- edge-typed message passing with max aggregation -------------------
    k_msg<<<cdiv(E, 64), 256, 0, stream>>>(feats, WfPk, bfb, src, dst, agg, E);
    k_msg<<<cdiv(E, 64), 256, 0, stream>>>(feats, WbPk, bbb, dst, src, agg, E);
    k_msg<<<cdiv(N, 64), 256, 0, stream>>>(feats, WsPk, bsb, nullptr, nullptr, agg, N);

    // --- result = agg @ Wout + bout ----------------------------------------
    k_result<<<cdiv(N, 64), 256, 0, stream>>>(agg, WoutPk, bout, resultbf, N);

    // --- MLP heads ----------------------------------------------------------
    float* outp = (float*)d_out;
    for (int t = 0; t < 2; ++t) {
        k_head<<<cdiv(KT, 64), 256, 0, stream>>>(resultbf, tgt_task, Wt1Pk[t], bt1 + t * 256,
                                                 Wt2Pk[t], bt2 + t * 2,
                                                 outp + (size_t)t * KT * 2, KT);
        k_head<<<cdiv(KM, 64), 256, 0, stream>>>(resultbf, tgt_map, Wm1Pk[t], bm1 + t * 256,
                                                 Wm2Pk[t], bm2 + t * 2,
                                                 outp + (size_t)2 * KT * 2 + (size_t)t * KM * 2, KM);
    }
}
